// GroupAttention_14499809591828
// MI455X (gfx1250) — matrix operand order repaired and checked by reading, not compiled, not run on hardware
//
#include <hip/hip_runtime.h>
#include <hip/hip_bf16.h>
#include <math.h>

// ---------------------------------------------------------------------------
// GroupAttention for MI455X (gfx1250): bf16 WMMA GEMMs + fused flash attention
//   B=16, S=1024, D=768, H=8, DK=96
//   Projections: 64x128 block tiles, 4 WMMA/wave/K-step.
//   Attention:   flash-style, double-buffered async global->LDS staging.
// ---------------------------------------------------------------------------

typedef __attribute__((ext_vector_type(16))) __bf16 v16bf;
typedef __attribute__((ext_vector_type(8)))  float  v8f;
typedef int v4i_gcc __attribute__((vector_size(16)));   // matches builtin param

#define BB 16
#define SS 1024
#define DD 768
#define HH 8
#define DK 96

union Frag16 { uint4 u4[2]; v16bf v; };

__device__ __forceinline__ unsigned short f2bf(float x) {
    unsigned int u = __float_as_uint(x);
    unsigned int r = u + 0x7FFFu + ((u >> 16) & 1u);   // round-to-nearest-even
    return (unsigned short)(r >> 16);
}

// A-matrix fragment (16x32 bf16, M=lane&15): interleaved K groups of 8
__device__ __forceinline__ v16bf load_fragA(const unsigned short* rowp, int lh) {
    Frag16 f;
    f.u4[0] = *(const uint4*)(rowp + lh * 8);
    f.u4[1] = *(const uint4*)(rowp + 16 + lh * 8);
    return f.v;
}

// B-matrix fragment (32x16 bf16, N=lane&15): interleaved K groups of 8
__device__ __forceinline__ v16bf load_fragB(const unsigned short* rowp, int lh) {
    Frag16 f;
    f.u4[0] = *(const uint4*)(rowp + lh * 8);
    f.u4[1] = *(const uint4*)(rowp + 16 + lh * 8);
    return f.v;
}

__device__ __forceinline__ v8f wmma_bf16(v16bf a, v16bf b, v8f c) {
    return __builtin_amdgcn_wmma_f32_16x16x32_bf16(
        false, a, false, b, (short)0, c, false, false);
}

// 16-byte async copy global -> LDS (ASYNCcnt path); VGPR-copy fallback.
__device__ __forceinline__ void async_copy16(const unsigned short* gsrc,
                                             unsigned short* ldst) {
#if defined(__gfx1250__) && __has_builtin(__builtin_amdgcn_global_load_async_to_lds_b128)
    __builtin_amdgcn_global_load_async_to_lds_b128(
        (__attribute__((address_space(1))) v4i_gcc*)gsrc,
        (__attribute__((address_space(3))) v4i_gcc*)ldst, 0, 0);
#else
    *(uint4*)ldst = *(const uint4*)gsrc;
#endif
}

__device__ __forceinline__ void async_wait_all() {
#if defined(__gfx1250__) && __has_builtin(__builtin_amdgcn_s_wait_asynccnt)
    __builtin_amdgcn_s_wait_asynccnt(0);
#elif defined(__gfx1250__)
    asm volatile("s_wait_asynccnt 0x0" ::: "memory");
#endif
}

// ---------------------------------------------------------------------------
// Y[M,N] = X[M,K] @ W[N,K]^T + bias    (M=16384, N=K=768)
// Block = 256 threads (8 waves, 2x4), block tile 64x128, wave tile 32x32.
// OUT_MODE: 0 = f32 row-major, 1 = bf16 row-major, 2 = bf16 [b][h][d][s]
// ---------------------------------------------------------------------------
template <bool IN_BF16, int OUT_MODE>
__global__ __launch_bounds__(256) void proj_gemm(
    const void* __restrict__ Xin, const float* __restrict__ W,
    const float* __restrict__ bias, void* __restrict__ Yout,
    int M, int N, int K)
{
    __shared__ unsigned short Als[64 * 40];    // 64x32 tile, 80B rows
    __shared__ unsigned short Bls[128 * 40];   // 128x32 tile

    const int tid  = threadIdx.x;
    const int wid  = tid >> 5;
    const int lane = tid & 31;
    const int lh   = lane >> 4;
    const int ln   = lane & 15;
    const int wm   = wid >> 2;          // 0..1
    const int wn   = wid & 3;           // 0..3
    const int m0   = blockIdx.x * 64;
    const int n0   = blockIdx.y * 128;

    v8f zero = {0.f, 0.f, 0.f, 0.f, 0.f, 0.f, 0.f, 0.f};
    v8f acc[2][2];
    acc[0][0] = zero; acc[0][1] = zero; acc[1][0] = zero; acc[1][1] = zero;

    for (int kt = 0; kt < K / 32; ++kt) {
        const int k0 = kt * 32;
        __syncthreads();
        // ---- stage A tile (64x32) as bf16 ----
        if (IN_BF16) {
            int row = tid >> 2, c = (tid & 3) * 8;
            const unsigned short* src =
                (const unsigned short*)Xin + (size_t)(m0 + row) * K + k0 + c;
            *(uint4*)&Als[row * 40 + c] = *(const uint4*)src;
        } else {
            for (int i = tid; i < 512; i += 256) {
                int row = i >> 3, c = (i & 7) * 4;
                const float* src = (const float*)Xin + (size_t)(m0 + row) * K + k0 + c;
                float4 f = *(const float4*)src;
                uint2 pk;
                pk.x = (unsigned)f2bf(f.x) | ((unsigned)f2bf(f.y) << 16);
                pk.y = (unsigned)f2bf(f.z) | ((unsigned)f2bf(f.w) << 16);
                *(uint2*)&Als[row * 40 + c] = pk;
            }
        }
        // ---- stage B tile (128x32 from W rows) as bf16 ----
        for (int i = tid; i < 1024; i += 256) {
            int row = i >> 3, c = (i & 7) * 4;
            const float* src = W + (size_t)(n0 + row) * K + k0 + c;
            float4 f = *(const float4*)src;
            uint2 pk;
            pk.x = (unsigned)f2bf(f.x) | ((unsigned)f2bf(f.y) << 16);
            pk.y = (unsigned)f2bf(f.z) | ((unsigned)f2bf(f.w) << 16);
            *(uint2*)&Bls[row * 40 + c] = pk;
        }
        __syncthreads();

        v16bf a0 = load_fragA(&Als[(wm * 32 + ln) * 40], lh);
        v16bf a1 = load_fragA(&Als[(wm * 32 + 16 + ln) * 40], lh);
        v16bf b0 = load_fragB(&Bls[(wn * 32 + ln) * 40], lh);
        v16bf b1 = load_fragB(&Bls[(wn * 32 + 16 + ln) * 40], lh);
        acc[0][0] = wmma_bf16(a0, b0, acc[0][0]);
        acc[0][1] = wmma_bf16(a0, b1, acc[0][1]);
        acc[1][0] = wmma_bf16(a1, b0, acc[1][0]);
        acc[1][1] = wmma_bf16(a1, b1, acc[1][1]);
    }

#pragma unroll
    for (int i = 0; i < 2; ++i) {
#pragma unroll
        for (int j = 0; j < 2; ++j) {
            const int nglob = n0 + wn * 32 + j * 16 + ln;
            const float bv  = bias[nglob];
            const int mbase = m0 + wm * 32 + i * 16 + lh * 8;
#pragma unroll
            for (int v = 0; v < 8; ++v) {
                float val = acc[i][j][v] + bv;
                if (OUT_MODE == 0) {
                    ((float*)Yout)[(size_t)(mbase + v) * N + nglob] = val;
                } else if (OUT_MODE == 1) {
                    ((unsigned short*)Yout)[(size_t)(mbase + v) * N + nglob] = f2bf(val);
                } else {                       // per-head transposed: [b][h][d][s]
                    int mrow = mbase + v;
                    int bsel = mrow >> 10, s = mrow & (SS - 1);
                    int hsel = nglob / DK, d = nglob % DK;
                    size_t off = ((size_t)(bsel * HH + hsel) * DK + d) * SS + s;
                    ((unsigned short*)Yout)[off] = f2bf(val);
                }
            }
        }
    }
}

// ---------------------------------------------------------------------------
// Fused flash attention: one block per (b, h, 128 query rows); 8 waves,
// each wave owns 16 query rows. Key loop in blocks of 32, double-buffered
// async DMA staging of K / V^T tiles.
//   Qp, Kp : bf16 [b][s][h*96+d] row-major
//   VpT    : bf16 [b][h][d][s]   (written transposed by the V projection)
// ---------------------------------------------------------------------------
__global__ __launch_bounds__(256) void attn_kernel(
    const unsigned short* __restrict__ Qp, const unsigned short* __restrict__ Kp,
    const unsigned short* __restrict__ VpT, const int* __restrict__ mask,
    unsigned short* __restrict__ Xout)
{
    __shared__ unsigned short Qs[128 * 96];         // Q tile  [qrow][dim]
    __shared__ unsigned short Ks[2][32 * 104];      // K tile  [key][dim]
    __shared__ unsigned short VTs[2][96 * 40];      // V tileT [dim][key]
    __shared__ unsigned short Pls[8 * 16 * 40];     // per-wave P tile [row][key]

    const int tid  = threadIdx.x;
    const int wid  = tid >> 5;
    const int lane = tid & 31;
    const int lh   = lane >> 4;
    const int ln   = lane & 15;

    const int q0 = blockIdx.x * 128;
    const int b  = blockIdx.y >> 3;      // /H
    const int h  = blockIdx.y & 7;       // %H
    const int g  = h & 3;                // concat(mask,mask) -> group = h%4
    const float scale = 0.10206207261596575f;    // 1/sqrt(96)

    const unsigned short* Kbase = Kp  + ((size_t)b * SS) * DD + h * DK;
    const unsigned short* Vbase = VpT + ((size_t)(b * HH + h) * DK) * SS;

    // issue one key-block's async staging into buffer `buf`
    auto stage_kv = [&](int kb, int buf) {
        for (int i = tid; i < 768; i += 256) {
            if (i < 384) {                 // K: 32 rows x 12 xb128
                int row = i / 12, c = i % 12;
                async_copy16(Kbase + (size_t)(kb * 32 + row) * DD + c * 8,
                             &Ks[buf][row * 104 + c * 8]);
            } else {                       // V^T: 96 rows x 4 xb128
                int j = i - 384;
                int d = j >> 2, c = j & 3;
                async_copy16(Vbase + (size_t)d * SS + kb * 32 + c * 8,
                             &VTs[buf][d * 40 + c * 8]);
            }
        }
    };

    // ---- stage Q block (128 x 96 bf16) + first K/V block via async DMA ----
    for (int i = tid; i < 128 * 12; i += 256) {
        int row = i / 12, c = i % 12;
        const unsigned short* src =
            Qp + ((size_t)b * SS + q0 + row) * DD + h * DK + c * 8;
        async_copy16(src, &Qs[row * 96 + c * 8]);
    }
    stage_kv(0, 0);
    async_wait_all();
    __syncthreads();

    v16bf qf[3];
#pragma unroll
    for (int kk = 0; kk < 3; ++kk)
        qf[kk] = load_fragA(&Qs[(wid * 16 + ln) * 96 + kk * 32], lh);

    v8f zero = {0.f, 0.f, 0.f, 0.f, 0.f, 0.f, 0.f, 0.f};
    v8f acc[6];
#pragma unroll
    for (int n = 0; n < 6; ++n) acc[n] = zero;
    float Mrow[8], Lrow[8];
#pragma unroll
    for (int v = 0; v < 8; ++v) { Mrow[v] = -INFINITY; Lrow[v] = 0.f; }

    for (int kb = 0; kb < SS / 32; ++kb) {
        const int cur = kb & 1;
        // prefetch next key block into the other buffer (overlaps compute)
        if (kb + 1 < SS / 32) stage_kv(kb + 1, cur ^ 1);

        // ---- scores: S = Q (16x96) x K^T (96x32) -> two 16x16 tiles ----
        v16bf kb0[3], kb1[3];
#pragma unroll
        for (int kk = 0; kk < 3; ++kk) {
            kb0[kk] = load_fragB(&Ks[cur][(0  + ln) * 104 + kk * 32], lh);
            kb1[kk] = load_fragB(&Ks[cur][(16 + ln) * 104 + kk * 32], lh);
        }
        v8f s0 = zero, s1 = zero;
#pragma unroll
        for (int kk = 0; kk < 3; ++kk) {
            s0 = wmma_bf16(qf[kk], kb0[kk], s0);
            s1 = wmma_bf16(qf[kk], kb1[kk], s1);
        }

        // ---- mask + online softmax (C layout: row = v + 8*lh, col = ln) ----
        unsigned short* pbase = &Pls[wid * 640];
#pragma unroll
        for (int v = 0; v < 8; ++v) {
            const int row = q0 + wid * 16 + v + 8 * lh;
            const int* mp = mask + (((size_t)(b * 4 + g) * SS + row) * SS)
                                 + kb * 32 + ln;
            float x0 = s0[v] * scale; if (mp[0]  == 0) x0 = -1e9f;
            float x1 = s1[v] * scale; if (mp[16] == 0) x1 = -1e9f;
            float rm = fmaxf(x0, x1);
            rm = fmaxf(rm, __shfl_xor(rm, 1, 32));
            rm = fmaxf(rm, __shfl_xor(rm, 2, 32));
            rm = fmaxf(rm, __shfl_xor(rm, 4, 32));
            rm = fmaxf(rm, __shfl_xor(rm, 8, 32));
            float Mn = fmaxf(Mrow[v], rm);
            float alpha = __expf(Mrow[v] - Mn);
            Mrow[v] = Mn;
            float p0 = __expf(x0 - Mn);
            float p1 = __expf(x1 - Mn);
            float ps = p0 + p1;
            ps += __shfl_xor(ps, 1, 32);
            ps += __shfl_xor(ps, 2, 32);
            ps += __shfl_xor(ps, 4, 32);
            ps += __shfl_xor(ps, 8, 32);
            Lrow[v] = Lrow[v] * alpha + ps;
#pragma unroll
            for (int n = 0; n < 6; ++n) acc[n][v] *= alpha;
            unsigned short* pr = pbase + (v + 8 * lh) * 40 + ln;
            pr[0]  = f2bf(p0);
            pr[16] = f2bf(p1);
        }
        // wave-local LDS RAW: P stores must land before fragment loads
        asm volatile("s_wait_dscnt 0" ::: "memory");

        // ---- O += P (16x32) x V (32x96) ----
        v16bf pa = load_fragA(&Pls[wid * 640 + ln * 40], lh);
        v16bf vb[6];
#pragma unroll
        for (int n = 0; n < 6; ++n)
            vb[n] = load_fragB(&VTs[cur][(n * 16 + ln) * 40], lh);
#pragma unroll
        for (int n = 0; n < 6; ++n)
            acc[n] = wmma_bf16(pa, vb[n], acc[n]);

        // next buffer must be fully landed before the next iteration reads it
        async_wait_all();
        __syncthreads();
    }

    // ---- normalize and write bf16 output [b][row][h*96 + dim] ----
#pragma unroll
    for (int v = 0; v < 8; ++v) {
        float inv = 1.0f / Lrow[v];
        const int row = q0 + wid * 16 + v + 8 * lh;
        unsigned short* op = Xout + ((size_t)b * SS + row) * DD + h * DK + ln;
#pragma unroll
        for (int n = 0; n < 6; ++n) op[n * 16] = f2bf(acc[n][v] * inv);
    }
}

// ---------------------------------------------------------------------------
extern "C" void kernel_launch(void* const* d_in, const int* in_sizes, int n_in,
                              void* d_out, int out_size, void* d_ws, size_t ws_size,
                              hipStream_t stream) {
    (void)in_sizes; (void)n_in; (void)out_size; (void)ws_size;

    const float* query = (const float*)d_in[0];
    const float* key   = (const float*)d_in[1];
    const float* value = (const float*)d_in[2];
    const int*   mask  = (const int*)d_in[3];
    const float* Wq = (const float*)d_in[4];  const float* bq = (const float*)d_in[5];
    const float* Wk = (const float*)d_in[6];  const float* bk = (const float*)d_in[7];
    const float* Wv = (const float*)d_in[8];  const float* bv = (const float*)d_in[9];
    const float* Wo = (const float*)d_in[10]; const float* bo = (const float*)d_in[11];
    float* out = (float*)d_out;

    const size_t elems = (size_t)BB * SS * DD;      // 12.58M
    unsigned short* Qp  = (unsigned short*)d_ws;    // bf16 intermediates
    unsigned short* Kp  = Qp + elems;
    unsigned short* VpT = Kp + elems;               // [b][h][d][s]
    unsigned short* Xa  = VpT + elems;              // total ~100 MB

    const int M = BB * SS, N = DD, K = DD;
    dim3 blk(256);
    dim3 pg(M / 64, N / 128);                       // 256 x 6

    proj_gemm<false, 1><<<pg, blk, 0, stream>>>(query, Wq, bq, Qp,  M, N, K);
    proj_gemm<false, 1><<<pg, blk, 0, stream>>>(key,   Wk, bk, Kp,  M, N, K);
    proj_gemm<false, 2><<<pg, blk, 0, stream>>>(value, Wv, bv, VpT, M, N, K);

    attn_kernel<<<dim3(SS / 128, BB * HH), blk, 0, stream>>>(Qp, Kp, VpT, mask, Xa);

    proj_gemm<true, 0><<<pg, blk, 0, stream>>>(Xa, Wo, bo, out, M, N, K);
}
